// FP4DiTLinear_37477884625347
// MI455X (gfx1250) — matrix operand order repaired and checked by reading, not compiled, not run on hardware
//
#include <hip/hip_runtime.h>
#include <hip/hip_bf16.h>
#include <stdint.h>

typedef __attribute__((ext_vector_type(16))) _Float16 v16h;
typedef __attribute__((ext_vector_type(8)))  _Float16 v8h;
typedef __attribute__((ext_vector_type(4)))  _Float16 v4h;
typedef __attribute__((ext_vector_type(8)))  float    v8f;
typedef __attribute__((ext_vector_type(4)))  float    v4f;
typedef __attribute__((ext_vector_type(4)))  int      v4i;

#if __has_builtin(__builtin_amdgcn_global_load_async_to_lds_b128) && \
    __has_builtin(__builtin_amdgcn_s_wait_asynccnt)
#define USE_ASYNC_LDS 1
#else
#define USE_ASYNC_LDS 0
#endif

// ---------------------------------------------------------------------------
// NVFP4 nearest-level rounding. Levels {0, .5, 1, 1.5, 2, 3, 4, 6} with
// midpoint thresholds (argmin over |t - level|).
// ---------------------------------------------------------------------------
__device__ __forceinline__ float nvfp4_level(float t) {
    float l = 0.0f;
    l = (t >= 0.25f) ? 0.5f : l;
    l = (t >= 0.75f) ? 1.0f : l;
    l = (t >= 1.25f) ? 1.5f : l;
    l = (t >= 1.75f) ? 2.0f : l;
    l = (t >= 2.50f) ? 3.0f : l;
    l = (t >= 3.50f) ? 4.0f : l;
    l = (t >= 5.00f) ? 6.0f : l;
    return l;
}

// ---------------------------------------------------------------------------
// Kernel 1: SmoothQuant prescale + blockwise NVFP4 fake quant -> f16.
// Per-token pre_scale cancels algebraically; each 16-elem block independent:
// out = sign(v) * level(|v|*6/amax) * amax/6,  v = x*smooth.
// ---------------------------------------------------------------------------
__global__ __launch_bounds__(256) void FP4DiT_quant_kernel(
    const float* __restrict__ x, const float* __restrict__ smooth,
    _Float16* __restrict__ xq, int nblk, int blocksPerRow) {
    int b0 = blockIdx.x * blockDim.x + threadIdx.x;
    int stride = gridDim.x * blockDim.x;
    for (int b = b0; b < nblk; b += stride) {
        const v4f* xp = (const v4f*)(x + (size_t)b * 16);
        const v4f* sp = (const v4f*)(smooth + (size_t)(b % blocksPerRow) * 16);
        float v[16];
        float amax = 0.0f;
#pragma unroll
        for (int q = 0; q < 4; ++q) {
            v4f xv = xp[q];
            v4f sv = sp[q];
#pragma unroll
            for (int j = 0; j < 4; ++j) {
                float t = xv[j] * sv[j];
                v[q * 4 + j] = t;
                amax = fmaxf(amax, fabsf(t));
            }
        }
        amax = fmaxf(amax, 1e-12f);
        float bs  = amax * (1.0f / 6.0f);
        float inv = 6.0f / amax;
        v8h r0, r1;
#pragma unroll
        for (int j = 0; j < 8; ++j) {
            float l0 = nvfp4_level(fabsf(v[j]) * inv);
            float l1 = nvfp4_level(fabsf(v[j + 8]) * inv);
            r0[j] = (_Float16)copysignf(l0 * bs, v[j]);
            r1[j] = (_Float16)copysignf(l1 * bs, v[j + 8]);
        }
        v8h* op = (v8h*)(xq + (size_t)b * 16);
        op[0] = r0;
        op[1] = r1;
    }
}

// ---------------------------------------------------------------------------
// Kernel 2: weights f32 -> f16.
// ---------------------------------------------------------------------------
__global__ __launch_bounds__(256) void FP4DiT_wconv_kernel(
    const float* __restrict__ w, _Float16* __restrict__ wh, size_t n4) {
    size_t i = (size_t)blockIdx.x * blockDim.x + threadIdx.x;
    size_t stride = (size_t)gridDim.x * blockDim.x;
    for (size_t q = i; q < n4; q += stride) {
        v4f t = *(const v4f*)(w + q * 4);
        v4h h;
        h[0] = (_Float16)t[0]; h[1] = (_Float16)t[1];
        h[2] = (_Float16)t[2]; h[3] = (_Float16)t[3];
        *(v4h*)(wh + q * 4) = h;
    }
}

// ---------------------------------------------------------------------------
// Kernel 3: C = A(MxK,f16) * Bw(NxK,f16)^T + bias, f32 out.
// Workgroup tile 64x128, TK=64. 8 wave32 waves in a 2(M)x4(N) grid; each
// wave owns 32x32 of C (4 v8f accumulators) -> 8 WMMAs per K-chunk with
// 2x fragment reuse. Double-buffered LDS (48 KB). Staging uses
// GLOBAL_LOAD_ASYNC_TO_LDS_B128 (ASYNCcnt) when available, else
// global_load_b128 + ds_store_b128.
// ---------------------------------------------------------------------------
#define TM 64
#define TN 128
#define TK 64

union frag_u { v16h v; v8h h[2]; };

__device__ __forceinline__ void stage_chunk(const _Float16* __restrict__ g,
                                            _Float16* lds) {
#if USE_ASYNC_LDS
    __builtin_amdgcn_global_load_async_to_lds_b128(
        (__attribute__((address_space(1))) v4i*)g,
        (__attribute__((address_space(3))) v4i*)lds,
        0, 0);
#else
    *(v8h*)lds = *(const v8h*)g;
#endif
}

__global__ __launch_bounds__(256, 1) void FP4DiT_gemm_kernel(
    const _Float16* __restrict__ A, const _Float16* __restrict__ Bw,
    const float* __restrict__ bias, float* __restrict__ C,
    int M, int N, int K) {
    __shared__ __align__(128) _Float16 sA[2][TM][TK];
    __shared__ __align__(128) _Float16 sB[2][TN][TK];

    const int tid  = threadIdx.x;
    const int lane = tid & 31;
    const int wave = tid >> 5;
    const int wr   = wave >> 2;      // 0..1 : M offset wr*32
    const int wc   = wave & 3;       // 0..3 : N offset wc*32
    const int m_   = lane & 15;
    const int hi   = lane >> 4;

    const int bm = blockIdx.y * TM;
    const int bn = blockIdx.x * TN;

    // Staging map: 16B chunks; A tile 64x64h = 512 chunks (2/thread),
    // B tile 128x64h = 1024 chunks (4/thread). row=c>>3, seg=(c&7)*8.
    const int ar0 = tid >> 3, as0 = (tid & 7) * 8;            // c = tid
    const int ar1 = (tid + 256) >> 3, as1 = as0;              // c = tid+256

    v8f acc[2][2];
#pragma unroll
    for (int i = 0; i < 2; ++i)
#pragma unroll
        for (int j = 0; j < 2; ++j) acc[i][j] = (v8f){};

    const int nk = K / TK;

    // ---- stage tile 0 ----
    {
        stage_chunk(A + (size_t)(bm + ar0) * K + as0, &sA[0][ar0][as0]);
        stage_chunk(A + (size_t)(bm + ar1) * K + as1, &sA[0][ar1][as1]);
#pragma unroll
        for (int i = 0; i < 4; ++i) {
            int c = tid + 256 * i;
            int r = c >> 3, s = (c & 7) * 8;
            stage_chunk(Bw + (size_t)(bn + r) * K + s, &sB[0][r][s]);
        }
    }
#if USE_ASYNC_LDS
    __builtin_amdgcn_s_wait_asynccnt(0);
#endif
    __syncthreads();

    for (int t = 0; t < nk; ++t) {
        const int cur = t & 1;
        if (t + 1 < nk) {
            const int k0 = (t + 1) * TK;
            stage_chunk(A + (size_t)(bm + ar0) * K + k0 + as0, &sA[cur ^ 1][ar0][as0]);
            stage_chunk(A + (size_t)(bm + ar1) * K + k0 + as1, &sA[cur ^ 1][ar1][as1]);
#pragma unroll
            for (int i = 0; i < 4; ++i) {
                int c = tid + 256 * i;
                int r = c >> 3, s = (c & 7) * 8;
                stage_chunk(Bw + (size_t)(bn + r) * K + k0 + s, &sB[cur ^ 1][r][s]);
            }
        }

        // ---- compute: 2 sub-steps of K=32, 4 WMMAs each ----
#pragma unroll
        for (int ks = 0; ks < TK; ks += 32) {
            frag_u a0, a1;
            const _Float16* pa0 = &sA[cur][wr * 32      + m_][ks + hi * 8];
            const _Float16* pa1 = &sA[cur][wr * 32 + 16 + m_][ks + hi * 8];
            a0.h[0] = *(const v8h*)(pa0);
            a0.h[1] = *(const v8h*)(pa0 + 16);
            a1.h[0] = *(const v8h*)(pa1);
            a1.h[1] = *(const v8h*)(pa1 + 16);

            const _Float16* pb0 = &sB[cur][wc * 32      + m_][ks + hi * 8];
            const _Float16* pb1 = &sB[cur][wc * 32 + 16 + m_][ks + hi * 8];
            v16h b0 = __builtin_shufflevector(*(const v8h*)pb0, *(const v8h*)(pb0 + 16), 0,1,2,3,4,5,6,7,8,9,10,11,12,13,14,15);
            v16h b1 = __builtin_shufflevector(*(const v8h*)pb1, *(const v8h*)(pb1 + 16), 0,1,2,3,4,5,6,7,8,9,10,11,12,13,14,15);

            acc[0][0] = __builtin_amdgcn_wmma_f32_16x16x32_f16(
                            false, a0.v, false, b0, (short)0, acc[0][0], false, false);
            acc[0][1] = __builtin_amdgcn_wmma_f32_16x16x32_f16(
                            false, a0.v, false, b1, (short)0, acc[0][1], false, false);
            acc[1][0] = __builtin_amdgcn_wmma_f32_16x16x32_f16(
                            false, a1.v, false, b0, (short)0, acc[1][0], false, false);
            acc[1][1] = __builtin_amdgcn_wmma_f32_16x16x32_f16(
                            false, a1.v, false, b1, (short)0, acc[1][1], false, false);
        }

#if USE_ASYNC_LDS
        __builtin_amdgcn_s_wait_asynccnt(0);
#endif
        __syncthreads();
    }

    // ---- epilogue: C/D layout — VGPR r: lanes0-15 M=r, lanes16-31 M=8+r ----
#pragma unroll
    for (int fn = 0; fn < 2; ++fn) {
        const int   n0 = bn + wc * 32 + fn * 16 + m_;
        const float bv = bias[n0];
#pragma unroll
        for (int fm = 0; fm < 2; ++fm) {
            const int m0 = bm + wr * 32 + fm * 16 + hi * 8;
#pragma unroll
            for (int r = 0; r < 8; ++r) {
                C[(size_t)(m0 + r) * N + n0] = acc[fm][fn][r] + bv;
            }
        }
    }
}

// ---------------------------------------------------------------------------
// Launch.  inputs: x[T*H] f32, w[O*H] f32, smooth[H] f32, bias[O] f32
// output: [T*O] f32
// ---------------------------------------------------------------------------
extern "C" void kernel_launch(void* const* d_in, const int* in_sizes, int n_in,
                              void* d_out, int out_size, void* d_ws, size_t ws_size,
                              hipStream_t stream) {
    const float* x      = (const float*)d_in[0];
    const float* w      = (const float*)d_in[1];
    const float* smooth = (const float*)d_in[2];
    const float* bias   = (const float*)d_in[3];
    float*       out    = (float*)d_out;

    const int H = in_sizes[2];          // 1152
    const int O = in_sizes[3];          // 4608
    const int T = in_sizes[0] / H;      // 16384
    const int M = T, N = O, K = H;

    _Float16* xq = (_Float16*)d_ws;
    size_t xq_bytes = (((size_t)M * K * sizeof(_Float16)) + 255) & ~(size_t)255;
    _Float16* wh = (_Float16*)((char*)d_ws + xq_bytes);

    const int blocksPerRow = K / 16;
    const int nblk = M * blocksPerRow;
    int qgrid = (nblk + 255) / 256;
    FP4DiT_quant_kernel<<<qgrid, 256, 0, stream>>>(x, smooth, xq, nblk, blocksPerRow);

    const size_t wn4 = ((size_t)N * K) / 4;
    int wgrid = (int)((wn4 + 255) / 256);
    FP4DiT_wconv_kernel<<<wgrid, 256, 0, stream>>>(w, wh, wn4);

    dim3 grid(N / TN, M / TM);   // 36 x 256
    FP4DiT_gemm_kernel<<<grid, 256, 0, stream>>>(xq, wh, bias, out, M, N, K);
}
